// MixtralSparseMoeBlock_4398046511747
// MI455X (gfx1250) — matrix operand order repaired and checked by reading, not compiled, not run on hardware
//
#include <hip/hip_runtime.h>
#include <math.h>

// ---------------------------------------------------------------------------
// Mixtral sparse MoE block for gfx1250 (MI455X), wave32 + WMMA bf16 + TDM.
//   T=4096 tokens, D=1024, F=3584, E=8 experts, top-2 routing.
// Pipeline:
//   1) fp32->bf16 pre-convert of x, w1, w3, w2 (weights then fit in 192MB L2)
//   2) router (wave/token, shuffle reduce, top-2 softmax, atomic compaction)
//   3) GEMM1: h = silu(x w1^T) * (x w3^T), B tiles staged via TENSOR_LOAD_TO_LDS
//   4) GEMM2: out += gate * (h w2^T), atomic f32 scatter (2 commutative adds)
// ---------------------------------------------------------------------------

typedef __attribute__((ext_vector_type(16))) __bf16 v16bf;
typedef __attribute__((ext_vector_type(8)))  __bf16 bf16x8;
typedef __attribute__((ext_vector_type(8)))  float  v8f;
typedef __attribute__((ext_vector_type(4)))  unsigned int u32x4;
typedef __attribute__((ext_vector_type(8)))  int i32x8;
typedef __attribute__((ext_vector_type(4)))  int i32x4;

#define NUM_T 4096
#define DIM_D 1024
#define DIM_F 3584
#define NUM_E 8
#define TILE_M 128
#define TILE_N 64
#define TILE_K 32

// fp32 -> bf16 bits, round-to-nearest-even
__device__ __forceinline__ unsigned short f2bf_bits(float f) {
    unsigned int u = __builtin_bit_cast(unsigned int, f);
    u += 0x7FFFu + ((u >> 16) & 1u);
    return (unsigned short)(u >> 16);
}
__device__ __forceinline__ __bf16 bits2bf(unsigned short s) {
    return __builtin_bit_cast(__bf16, s);
}
// low 32 bits of a flat shared pointer == LDS byte offset (aperture mapping)
__device__ __forceinline__ unsigned int lds_off(const void* p) {
    return (unsigned int)(unsigned long long)(size_t)p;
}

// ---------------------------------------------------------------------------
// TDM: async 2D tile load (rows x row_elems of bf16) global -> LDS.
// D# per CDNA5 ISA ch.8: group0 {count, lds_addr, global_addr, type=2},
// group1 {data_size=1(2B), tensor dims (no-OOB sentinels), tile dims, stride}.
// Tracked by TENSORcnt. Issue from ONE wave per workgroup (EXEC ignored).
// ---------------------------------------------------------------------------
__device__ __forceinline__ void tdm_load_2d_bf16(unsigned int lds_addr,
                                                 const void* gptr,
                                                 unsigned int rows,
                                                 unsigned int row_elems,
                                                 unsigned int row_stride_elems) {
    unsigned long long ga = (unsigned long long)(size_t)gptr;
    u32x4 g0;
    g0.x = 1u;                                    // count=1, user mode
    g0.y = lds_addr;                              // lds_addr [63:32]
    g0.z = (unsigned int)ga;                      // global_addr [95:64]
    g0.w = (unsigned int)((ga >> 32) & 0x1FFFFFFu) | (2u << 30); // [120:96]+type=2

    i32x8 g1;
    const unsigned int td0 = 0x7FFFFFFFu;         // tensor_dim0 sentinel (no OOB)
    const unsigned int td1 = 0x7FFFFFFFu;         // tensor_dim1 sentinel
    g1[0] = (int)(1u << 16);                      // data_size=1 (2 bytes)
    g1[1] = (int)((td0 & 0xFFFFu) << 16);         // [31:16]=td0 lo16
    g1[2] = (int)((td0 >> 16) | ((td1 & 0xFFFFu) << 16));
    g1[3] = (int)((td1 >> 16) | (row_elems << 16));   // [31:16]=tile_dim0
    g1[4] = (int)(rows & 0xFFFFu);                // tile_dim1, tile_dim2=0
    g1[5] = (int)row_stride_elems;                // tensor_dim0_stride lo32
    g1[6] = 0;                                    // stride hi16, dim1_stride lo16
    g1[7] = 0;

    i32x4 z4 = {0, 0, 0, 0};
    i32x8 z8 = {0, 0, 0, 0, 0, 0, 0, 0};
    __builtin_amdgcn_tensor_load_to_lds(g0, g1, z4, z4, z8, 0);
}

// ---------------------------------------------------------------------------
// fp32 -> bf16 bulk converter (vectorized: float4 in, 4x bf16 out)
// ---------------------------------------------------------------------------
__global__ void cvt_f32_bf16(const float4* __restrict__ src,
                             uint2* __restrict__ dst, int n4) {
    int i = blockIdx.x * blockDim.x + threadIdx.x;
    if (i >= n4) return;
    float4 v = src[i];
    unsigned int lo = ((unsigned int)f2bf_bits(v.y) << 16) | f2bf_bits(v.x);
    unsigned int hi = ((unsigned int)f2bf_bits(v.w) << 16) | f2bf_bits(v.z);
    dst[i] = make_uint2(lo, hi);
}

// ---------------------------------------------------------------------------
// Router: one wave per token; softmax->top2->renorm; atomic compaction.
// ---------------------------------------------------------------------------
__global__ void moe_router(const float* __restrict__ x,
                           const float* __restrict__ gw,
                           int* __restrict__ counts,
                           int* __restrict__ tok_list,
                           float* __restrict__ gate_list) {
    const int token = blockIdx.x * 8 + (threadIdx.x >> 5);
    const int lane  = threadIdx.x & 31;
    if (token >= NUM_T) return;
    const float* xr = x + (size_t)token * DIM_D;

    float logits[NUM_E];
#pragma unroll
    for (int e = 0; e < NUM_E; ++e) {
        float p = 0.f;
        const float* ge = gw + e * DIM_D;
        for (int k = lane; k < DIM_D; k += 32) p += xr[k] * ge[k];
#pragma unroll
        for (int off = 16; off > 0; off >>= 1) p += __shfl_xor(p, off, 32);
        logits[e] = p;
    }
    if (lane == 0) {
        int i1 = 0; float v1 = logits[0];
#pragma unroll
        for (int e = 1; e < NUM_E; ++e)
            if (logits[e] > v1) { v1 = logits[e]; i1 = e; }
        int i2 = -1; float v2 = -INFINITY;
#pragma unroll
        for (int e = 0; e < NUM_E; ++e)
            if (e != i1 && logits[e] > v2) { v2 = logits[e]; i2 = e; }
        const float e2 = __expf(v2 - v1);
        const float inv = 1.0f / (1.0f + e2);

        int s1 = atomicAdd(&counts[i1], 1);
        tok_list[i1 * NUM_T + s1]  = token;
        gate_list[i1 * NUM_T + s1] = inv;          // e1*inv, e1==1
        int s2 = atomicAdd(&counts[i2], 1);
        tok_list[i2 * NUM_T + s2]  = token;
        gate_list[i2 * NUM_T + s2] = e2 * inv;
    }
}

__global__ void moe_offsets(const int* __restrict__ counts, int* __restrict__ offsets) {
    if (threadIdx.x == 0) {
        int acc = 0;
#pragma unroll
        for (int e = 0; e < NUM_E; ++e) { offsets[e] = acc; acc += counts[e]; }
    }
}

// ---------------------------------------------------------------------------
// GEMM1: h[slot,:] = silu(x[tok] w1^T) * (x[tok] w3^T); all operands bf16.
// Block 256 thr (8 waves); tile 128(M) x 64(N over F); K-step 32 over D.
// B tiles (w1,w3) double-buffered in LDS via TDM issued by wave 0.
// All 8 B fragments are preloaded before the WMMA chain so the ds_loads
// batch under one dscnt wait instead of serializing per WMMA.
// ---------------------------------------------------------------------------
__global__ void __launch_bounds__(256)
moe_gemm1(const unsigned short* __restrict__ xb,
          const unsigned short* __restrict__ w1b,
          const unsigned short* __restrict__ w3b,
          const int*   __restrict__ counts,
          const int*   __restrict__ offsets,
          const int*   __restrict__ tok_list,
          unsigned short* __restrict__ h) {
    const int e  = blockIdx.z;
    const int Ne = counts[e];
    const int m0 = blockIdx.x * TILE_M;
    if (m0 >= Ne) return;
    const int n0 = blockIdx.y * TILE_N;            // offset into F

    __shared__ unsigned short Bt[2][2][TILE_N][TILE_K];   // [buf][w1|w3][n][k]

    const int tid  = threadIdx.x;
    const int lane = tid & 31;
    const int wave = tid >> 5;
    const int half = lane >> 4;
    const int l16  = lane & 15;

    const int aSlot  = m0 + wave * 16 + l16;
    const int token  = (aSlot < Ne) ? tok_list[e * NUM_T + aSlot] : 0;
    const __bf16* xr = (const __bf16*)(xb + (size_t)token * DIM_D);

    const unsigned short* w1p = w1b + (size_t)e * DIM_F * DIM_D + (size_t)n0 * DIM_D;
    const unsigned short* w3p = w3b + (size_t)e * DIM_F * DIM_D + (size_t)n0 * DIM_D;

    v8f acc1[4], acc3[4];
#pragma unroll
    for (int nb = 0; nb < 4; ++nb) { acc1[nb] = (v8f){}; acc3[nb] = (v8f){}; }

    // prologue: DMA first K tile into buffer 0
    if (wave == 0) {
        tdm_load_2d_bf16(lds_off(&Bt[0][0][0][0]), w1p, TILE_N, TILE_K, DIM_D);
        tdm_load_2d_bf16(lds_off(&Bt[0][1][0][0]), w3p, TILE_N, TILE_K, DIM_D);
        __builtin_amdgcn_s_wait_tensorcnt(0);
    }
    __syncthreads();

    int buf = 0;
    for (int k0 = 0; k0 < DIM_D; k0 += TILE_K) {
        // overlap: DMA next K tile into the other buffer while computing
        if (wave == 0 && k0 + TILE_K < DIM_D) {
            tdm_load_2d_bf16(lds_off(&Bt[buf ^ 1][0][0][0]), w1p + k0 + TILE_K,
                             TILE_N, TILE_K, DIM_D);
            tdm_load_2d_bf16(lds_off(&Bt[buf ^ 1][1][0][0]), w3p + k0 + TILE_K,
                             TILE_N, TILE_K, DIM_D);
        }

        // A fragment (bf16, 2x16B vector loads): K = half*8+[0..7], 16+half*8+[0..7]
        v16bf a;
        {
            bf16x8 lo = *(const bf16x8*)(xr + k0 + half * 8);
            bf16x8 hi = *(const bf16x8*)(xr + k0 + 16 + half * 8);
#pragma unroll
            for (int j = 0; j < 8; ++j) { a[j] = lo[j]; a[8 + j] = hi[j]; }
        }

        // preload ALL B fragments of this iteration, then run the WMMA chain
        v16bf b1[4], b3[4];
#pragma unroll
        for (int nb = 0; nb < 4; ++nb) {
            const int col = nb * 16 + l16;
            const int kb  = half * 8;
#pragma unroll
            for (int j = 0; j < 16; ++j) {
                b1[nb][j] = bits2bf(Bt[buf][0][col][kb + j + (j >= 8 ? 8 : 0)]);
                b3[nb][j] = bits2bf(Bt[buf][1][col][kb + j + (j >= 8 ? 8 : 0)]);
            }
        }
#pragma unroll
        for (int nb = 0; nb < 4; ++nb) {
            acc1[nb] = __builtin_amdgcn_wmma_f32_16x16x32_bf16(
                false, a, false, b1[nb], (short)0, acc1[nb], false, false);
            acc3[nb] = __builtin_amdgcn_wmma_f32_16x16x32_bf16(
                false, a, false, b3[nb], (short)0, acc3[nb], false, false);
        }

        if (wave == 0) __builtin_amdgcn_s_wait_tensorcnt(0);
        __syncthreads();
        buf ^= 1;
    }

    // epilogue: h = silu(acc1) * acc3 -> bf16 (fast v_rcp_f32 sigmoid)
    const size_t rowBase = (size_t)offsets[e];
#pragma unroll
    for (int r = 0; r < 8; ++r) {
        const int m    = r + 8 * half;             // C/D: row = vgpr + 8*(lane>>4)
        const int slot = m0 + wave * 16 + m;
        if (slot < Ne) {
            unsigned short* hr = h + (rowBase + slot) * (size_t)DIM_F + n0;
#pragma unroll
            for (int nb = 0; nb < 4; ++nb) {
                const float z   = acc1[nb][r];
                const float sig = __builtin_amdgcn_rcpf(1.0f + __expf(-z));
                hr[nb * 16 + l16] = f2bf_bits(z * sig * acc3[nb][r]);
            }
        }
    }
}

// ---------------------------------------------------------------------------
// GEMM2: out[tok,:] += gate * (h[slot,:] w2^T). K over F (112 steps), N over D.
// ---------------------------------------------------------------------------
__global__ void __launch_bounds__(256)
moe_gemm2(const unsigned short* __restrict__ h,
          const unsigned short* __restrict__ w2b,
          const int*   __restrict__ counts,
          const int*   __restrict__ offsets,
          const int*   __restrict__ tok_list,
          const float* __restrict__ gate_list,
          float* __restrict__ out) {
    const int e  = blockIdx.z;
    const int Ne = counts[e];
    const int m0 = blockIdx.x * TILE_M;
    if (m0 >= Ne) return;
    const int n0 = blockIdx.y * TILE_N;            // offset into D

    __shared__ unsigned short Bt[2][TILE_N][TILE_K];

    const int tid  = threadIdx.x;
    const int lane = tid & 31;
    const int wave = tid >> 5;
    const int half = lane >> 4;
    const int l16  = lane & 15;

    const int aSlot = m0 + wave * 16 + l16;
    const int aRow  = (aSlot < Ne) ? aSlot : 0;
    const __bf16* hr = (const __bf16*)(h + ((size_t)offsets[e] + aRow) * (size_t)DIM_F);

    const unsigned short* w2p = w2b + (size_t)e * DIM_D * DIM_F + (size_t)n0 * DIM_F;

    v8f acc[4];
#pragma unroll
    for (int nb = 0; nb < 4; ++nb) acc[nb] = (v8f){};

    if (wave == 0) {
        tdm_load_2d_bf16(lds_off(&Bt[0][0][0]), w2p, TILE_N, TILE_K, DIM_F);
        __builtin_amdgcn_s_wait_tensorcnt(0);
    }
    __syncthreads();

    int buf = 0;
    for (int k0 = 0; k0 < DIM_F; k0 += TILE_K) {
        if (wave == 0 && k0 + TILE_K < DIM_F) {
            tdm_load_2d_bf16(lds_off(&Bt[buf ^ 1][0][0]), w2p + k0 + TILE_K,
                             TILE_N, TILE_K, DIM_F);
        }

        v16bf a;
        {
            bf16x8 lo = *(const bf16x8*)(hr + k0 + half * 8);
            bf16x8 hi = *(const bf16x8*)(hr + k0 + 16 + half * 8);
#pragma unroll
            for (int j = 0; j < 8; ++j) { a[j] = lo[j]; a[8 + j] = hi[j]; }
        }

        v16bf b[4];
#pragma unroll
        for (int nb = 0; nb < 4; ++nb) {
            const int col = nb * 16 + l16;
            const int kb  = half * 8;
#pragma unroll
            for (int j = 0; j < 16; ++j) b[nb][j] = bits2bf(Bt[buf][col][kb + j + (j >= 8 ? 8 : 0)]);
        }
#pragma unroll
        for (int nb = 0; nb < 4; ++nb) {
            acc[nb] = __builtin_amdgcn_wmma_f32_16x16x32_bf16(
                false, a, false, b[nb], (short)0, acc[nb], false, false);
        }

        if (wave == 0) __builtin_amdgcn_s_wait_tensorcnt(0);
        __syncthreads();
        buf ^= 1;
    }

    // gate-scaled atomic scatter-add (exactly 2 commutative adds per element)
#pragma unroll
    for (int r = 0; r < 8; ++r) {
        const int m    = r + 8 * half;
        const int slot = m0 + wave * 16 + m;
        if (slot < Ne) {
            const int   tok = tok_list[e * NUM_T + slot];
            const float g   = gate_list[e * NUM_T + slot];
            float* orow = out + (size_t)tok * DIM_D + n0;
#pragma unroll
            for (int nb = 0; nb < 4; ++nb)
                atomicAdd(&orow[nb * 16 + l16], g * acc[nb][r]);
        }
    }
}

// ---------------------------------------------------------------------------
// Host launch (graph-capture safe: async memsets + kernel launches only)
// ---------------------------------------------------------------------------
extern "C" void kernel_launch(void* const* d_in, const int* in_sizes, int n_in,
                              void* d_out, int out_size, void* d_ws, size_t ws_size,
                              hipStream_t stream) {
    const float* x  = (const float*)d_in[0];   // [2,2048,1024]
    const float* gw = (const float*)d_in[1];   // [8,1024]
    const float* w1 = (const float*)d_in[2];   // [8,3584,1024]
    const float* w3 = (const float*)d_in[3];   // [8,3584,1024]
    const float* w2 = (const float*)d_in[4];   // [8,1024,3584]
    float* out = (float*)d_out;
    (void)in_sizes; (void)n_in; (void)ws_size;

    const size_t W_ELEMS = (size_t)NUM_E * DIM_F * DIM_D;   // 29,360,128
    const size_t X_ELEMS = (size_t)NUM_T * DIM_D;           //  4,194,304

    // workspace layout (~243 MB)
    char* ws = (char*)d_ws;
    int*   counts    = (int*)ws;                 ws += 256;
    int*   offsets   = (int*)ws;                 ws += 256;
    int*   tok_list  = (int*)ws;                 ws += (size_t)NUM_E * NUM_T * sizeof(int);
    float* gate_list = (float*)ws;               ws += (size_t)NUM_E * NUM_T * sizeof(float);
    unsigned short* xb  = (unsigned short*)ws;   ws += X_ELEMS * 2;
    unsigned short* w1b = (unsigned short*)ws;   ws += W_ELEMS * 2;
    unsigned short* w3b = (unsigned short*)ws;   ws += W_ELEMS * 2;
    unsigned short* w2b = (unsigned short*)ws;   ws += W_ELEMS * 2;
    unsigned short* hbuf = (unsigned short*)ws;  // 2T x F bf16 (~58.7 MB)

    (void)hipMemsetAsync(counts, 0, 256, stream);
    (void)hipMemsetAsync(tok_list, 0, (size_t)NUM_E * NUM_T * sizeof(int), stream);
    (void)hipMemsetAsync(d_out, 0, (size_t)out_size * sizeof(float), stream);

    // bulk fp32 -> bf16 conversions
    {
        const int n4w = (int)(W_ELEMS / 4), n4x = (int)(X_ELEMS / 4);
        cvt_f32_bf16<<<(n4x + 255) / 256, 256, 0, stream>>>((const float4*)x,  (uint2*)xb,  n4x);
        cvt_f32_bf16<<<(n4w + 255) / 256, 256, 0, stream>>>((const float4*)w1, (uint2*)w1b, n4w);
        cvt_f32_bf16<<<(n4w + 255) / 256, 256, 0, stream>>>((const float4*)w3, (uint2*)w3b, n4w);
        cvt_f32_bf16<<<(n4w + 255) / 256, 256, 0, stream>>>((const float4*)w2, (uint2*)w2b, n4w);
    }

    moe_router<<<NUM_T / 8, 256, 0, stream>>>(x, gw, counts, tok_list, gate_list);
    moe_offsets<<<1, 32, 0, stream>>>(counts, offsets);

    dim3 g1(NUM_T / TILE_M, DIM_F / TILE_N, NUM_E);   // 32 x 56 x 8
    moe_gemm1<<<g1, 256, 0, stream>>>(xb, w1b, w3b, counts, offsets, tok_list, hbuf);

    dim3 g2(NUM_T / TILE_M, DIM_D / TILE_N, NUM_E);   // 32 x 16 x 8
    moe_gemm2<<<g2, 256, 0, stream>>>(hbuf, w2b, counts, offsets, tok_list, gate_list, out);
}
